// ProHealth_VAE_67740224193034
// MI455X (gfx1250) — matrix operand order repaired and checked by reading, not compiled, not run on hardware
//
#include <hip/hip_runtime.h>
#include <hip/hip_bf16.h>

// ---------------------------------------------------------------------------
// Problem constants (from reference)
// ---------------------------------------------------------------------------
#define NB    128     // batch
#define TSEQ  64      // visit sequence length
#define EMBD  128     // embedding dim
#define HD    128     // hidden dim
#define NCLS  128     // classes
#define TOUT  8       // output timestamps
#define NSUB  10      // RK4 substeps per unit interval
#define DTs   0.1f
#define VD    6000
#define VM    4000
#define VP    2000

typedef __attribute__((ext_vector_type(16))) _Float16 v16h;
typedef __attribute__((ext_vector_type(8)))  _Float16 v8h;
typedef __attribute__((ext_vector_type(8)))  float    v8f;
typedef unsigned int u32x4 __attribute__((ext_vector_type(4)));
typedef int          i32x4 __attribute__((ext_vector_type(4)));
typedef int          i32x8 __attribute__((ext_vector_type(8)));

#if defined(__has_builtin)
#  if __has_builtin(__builtin_amdgcn_tensor_load_to_lds)
#    define HAVE_TDM 1
#  endif
#  if __has_builtin(__builtin_amdgcn_s_wait_tensorcnt)
#    define HAVE_TWAIT 1
#  endif
#endif
#ifndef HAVE_TDM
#  define HAVE_TDM 0
#endif
#ifndef HAVE_TWAIT
#  define HAVE_TWAIT 0
#endif

__device__ __forceinline__ void wait_tensor0() {
#if HAVE_TWAIT
  __builtin_amdgcn_s_wait_tensorcnt(0);
#else
  asm volatile("s_wait_tensorcnt 0x0" ::: "memory");
#endif
}

__device__ __forceinline__ v8h pack8(float a0, float a1, float a2, float a3,
                                     float a4, float a5, float a6, float a7) {
  v8h p;
  p[0] = (_Float16)a0; p[1] = (_Float16)a1;
  p[2] = (_Float16)a2; p[3] = (_Float16)a3;
  p[4] = (_Float16)a4; p[5] = (_Float16)a5;
  p[6] = (_Float16)a6; p[7] = (_Float16)a7;
  return p;
}

// ---------------------------------------------------------------------------
// Generic WMMA GEMM: C[r,c] = sum_k A[r,k]*B[k,c] (+bias[c]) (+ReLU)
//   A element at A + bz*a_bs + r*a_rs + k*a_cs   (fp32 -> f16)
//   B element at B + bz*b_bs + k*b_rs + c*b_cs   (fp32 -> f16)
//   C element at C + bz*c_bs + r*c_rs + c        (fp32 WMMA accumulate)
// Block: 256 threads = 8 waves; tile 64(M) x 128(N), K chunked by 32.
// LDS layouts are fragment-major so each lane fetches its WMMA operands
// with aligned 16-byte ds loads:
//   Asm  [m][K]  row-major  (A lane data = two contiguous K-runs of 8)
//   BsmT [n][K]  transposed (B lane data = two contiguous K-runs of 8)
// When B rows are contiguous (b_cs==1) the fp32 tile is DMA'd by the
// Tensor Data Mover into LDS and converted/transposed from there.
// ---------------------------------------------------------------------------
#define MT 64
#define NTL 128
#define KT 32
#define KP (KT + 8)   // 40 halves = 80B pitch (16B multiple)

__global__ void __launch_bounds__(256)
gemm_wmma(const float* __restrict__ A, long a_bs, int a_rs, int a_cs,
          const float* __restrict__ Bm, long b_bs, int b_rs, int b_cs,
          const float* __restrict__ bias, int bias_bs,
          float* __restrict__ C, long c_bs, int c_rs,
          int M, int N, int K, int act)
{
  __shared__ __align__(16) _Float16 Asm[MT][KP];
  __shared__ __align__(16) _Float16 BsmT[NTL][KP];
#if HAVE_TDM
  __shared__ __align__(16) float    Braw[KT][NTL];
#endif
  const int tid  = threadIdx.x;
  const int lane = tid & 31;
  const int wave = tid >> 5;
  const int wm   = wave & 3;          // 4 M-tiles of 16
  const int wn   = wave >> 2;         // 2 N-halves of 64
  const int bm = blockIdx.x, bn = blockIdx.y, bz = blockIdx.z;

  const float* Ab = A  + (long)bz * a_bs;
  const float* Bb = Bm + (long)bz * b_bs;
  float*       Cb = C  + (long)bz * c_bs;

  v8f acc[4] = {};

  for (int k0 = 0; k0 < K; k0 += KT) {
    // ---- stage A tile (64 x 32) as f16; one v8h (16B) LDS store/thread ----
    {
      int r = tid & 63, oct = tid >> 6;        // 64 rows x 4 K-octets = 256
      int gr = bm * MT + r;
      int gk0 = k0 + 8 * oct;
      const float* ap = Ab + (long)gr * a_rs + (long)gk0 * a_cs;
      v8h p;
      if (gr < M && gk0 + 8 <= K) {            // fast path: no per-elem guards
        if (a_cs == 1) {                       // contiguous K: 2x float4
          float4 q0 = *(const float4*)(ap);
          float4 q1 = *(const float4*)(ap + 4);
          p = pack8(q0.x, q0.y, q0.z, q0.w, q1.x, q1.y, q1.z, q1.w);
        } else {                               // strided K (lane-coalesced)
          float v0 = ap[0];
          float v1 = ap[(long)1 * a_cs];
          float v2 = ap[(long)2 * a_cs];
          float v3 = ap[(long)3 * a_cs];
          float v4 = ap[(long)4 * a_cs];
          float v5 = ap[(long)5 * a_cs];
          float v6 = ap[(long)6 * a_cs];
          float v7 = ap[(long)7 * a_cs];
          p = pack8(v0, v1, v2, v3, v4, v5, v6, v7);
        }
      } else {                                 // rare tail: guarded
#pragma unroll
        for (int i = 0; i < 8; ++i) {
          int gk = gk0 + i;
          float v = (gr < M && gk < K) ? Ab[(long)gr * a_rs + (long)gk * a_cs] : 0.f;
          p[i] = (_Float16)v;
        }
      }
      *(v8h*)&Asm[r][8 * oct] = p;
    }
    // ---- stage B tile (32 x 128) transposed as f16 ----
    bool full = (k0 + KT <= K) && ((bn * NTL + NTL) <= N);
#if HAVE_TDM
    if (b_cs == 1 && full) {
      // TDM: DMA 32 rows x 128 fp32 straight into LDS, then convert+transpose.
      if (tid == 0) {
        const float* gsrc = Bb + (long)k0 * b_rs + (long)bn * NTL;
        unsigned long long ga = (unsigned long long)(const void*)gsrc;
        unsigned lds = (unsigned)(unsigned long long)(void*)&Braw[0][0];
        u32x4 g0;
        g0.x = 1u;                                   // count=1, user desc
        g0.y = lds;                                  // lds_addr
        g0.z = (unsigned)(ga & 0xffffffffu);         // global_addr[31:0]
        g0.w = (unsigned)((ga >> 32) & 0x1ffffffu) | (2u << 30); // type=2
        i32x8 g1;
        g1[0] = (int)(2u << 16);                     // data_size = 4 bytes
        g1[1] = (int)((NTL & 0xffffu) << 16);        // tensor_dim0[15:0]
        g1[2] = (int)((((unsigned)NTL >> 16) & 0xffffu) | ((KT & 0xffffu) << 16));
        g1[3] = (int)((((unsigned)KT >> 16) & 0xffffu) | ((unsigned)NTL << 16)); // tile_dim0
        g1[4] = (int)(KT & 0xffffu);                 // tile_dim1 = 32 rows
        g1[5] = (int)b_rs;                           // tensor_dim0_stride (elems)
        g1[6] = 0; g1[7] = 0;
        i32x4 gz4 = {0, 0, 0, 0};
        i32x8 gz8 = {0, 0, 0, 0, 0, 0, 0, 0};
        __builtin_amdgcn_tensor_load_to_lds(g0, g1, gz4, gz4, gz8, 0);
        wait_tensor0();
      }
      __syncthreads();
      for (int idx = tid; idx < NTL * 4; idx += 256) {
        int n = idx & 127, oct = idx >> 7;
        v8h p;
#pragma unroll
        for (int i = 0; i < 8; ++i) p[i] = (_Float16)Braw[8 * oct + i][n];
        *(v8h*)&BsmT[n][8 * oct] = p;
      }
    } else
#endif
    {
      for (int idx = tid; idx < NTL * 4; idx += 256) {
        int n = idx & 127, oct = idx >> 7;
        int gn = bn * NTL + n;
        int gk0 = k0 + 8 * oct;
        const float* bp = Bb + (long)gk0 * b_rs + (long)gn * b_cs;
        v8h p;
        if (gn < N && gk0 + 8 <= K) {          // fast path
          if (b_rs == 1) {                     // contiguous K: 2x float4
            float4 q0 = *(const float4*)(bp);
            float4 q1 = *(const float4*)(bp + 4);
            p = pack8(q0.x, q0.y, q0.z, q0.w, q1.x, q1.y, q1.z, q1.w);
          } else {
            float v0 = bp[0];
            float v1 = bp[(long)1 * b_rs];
            float v2 = bp[(long)2 * b_rs];
            float v3 = bp[(long)3 * b_rs];
            float v4 = bp[(long)4 * b_rs];
            float v5 = bp[(long)5 * b_rs];
            float v6 = bp[(long)6 * b_rs];
            float v7 = bp[(long)7 * b_rs];
            p = pack8(v0, v1, v2, v3, v4, v5, v6, v7);
          }
        } else {                               // rare tail: guarded
#pragma unroll
          for (int i = 0; i < 8; ++i) {
            int gk = gk0 + i;
            float v = (gk < K && gn < N) ? Bb[(long)gk * b_rs + (long)gn * b_cs] : 0.f;
            p[i] = (_Float16)v;
          }
        }
        *(v8h*)&BsmT[n][8 * oct] = p;
      }
    }
    __syncthreads();

    // ---- A fragment: two aligned 16B LDS loads + register shuffle ----
    v16h af;
    {
      int m  = wm * 16 + (lane & 15);
      int kb = (lane < 16) ? 0 : 8;
      v8h alo = *(const v8h*)&Asm[m][kb];        // K = kb .. kb+7
      v8h ahi = *(const v8h*)&Asm[m][kb + 16];   // K = kb+16 .. kb+23
      af = __builtin_shufflevector(alo, ahi, 0, 1, 2, 3, 4, 5, 6, 7,
                                   8, 9, 10, 11, 12, 13, 14, 15);
    }
    // ---- B fragments: two 8-half K-runs per lane, same K order as A ----
    v16h bf[4];
#pragma unroll
    for (int nt = 0; nt < 4; ++nt) {
      int n  = wn * 64 + nt * 16 + (lane & 15);
      int kb = (lane < 16) ? 0 : 8;
      v8h blo = *(const v8h*)&BsmT[n][kb];       // K = kb .. kb+7
      v8h bhi = *(const v8h*)&BsmT[n][kb + 16];  // K = kb+16 .. kb+23
      bf[nt] = __builtin_shufflevector(blo, bhi, 0, 1, 2, 3, 4, 5, 6, 7,
                                       8, 9, 10, 11, 12, 13, 14, 15);
    }
#pragma unroll
    for (int nt = 0; nt < 4; ++nt)
      acc[nt] = __builtin_amdgcn_wmma_f32_16x16x32_f16(
          false, af, false, bf[nt], (short)0, acc[nt], false, false);
    __syncthreads();
  }

  // ---- epilogue: f32 C layout -> global ----
  int mrow = wm * 16 + ((lane < 16) ? 0 : 8);
  int col0 = wn * 64 + (lane & 15);
#pragma unroll
  for (int nt = 0; nt < 4; ++nt) {
    int gc = bn * NTL + col0 + nt * 16;
    if (gc >= N) continue;
    float bv = bias ? bias[(long)bz * bias_bs + gc] : 0.f;
#pragma unroll
    for (int r = 0; r < 8; ++r) {
      int gr = bm * MT + mrow + r;
      if (gr >= M) continue;
      float v = acc[nt][r] + bv;
      if (act == 1) v = fmaxf(v, 0.f);
      Cb[(long)gr * c_rs + gc] = v;
    }
  }
}

// ---------------------------------------------------------------------------
// GRU time scan (sequential in T). One block per (b, modality), 128 threads.
// ---------------------------------------------------------------------------
__global__ void gru_scan_kernel(const float* __restrict__ gi,
                                const float* __restrict__ Whh,
                                const float* __restrict__ bhh,
                                float* __restrict__ hs)
{
  const int b = blockIdx.x, m = blockIdx.y, j = threadIdx.x;
  __shared__ float h[HD];
  h[j] = 0.f;
  __syncthreads();
  const float* W  = Whh + (long)m * 3 * HD * HD;
  const float* bh = bhh + m * 3 * HD;
  const float* g  = gi + ((long)m * NB + b) * TSEQ * (3 * HD);
  float*       o  = hs + ((long)m * NB + b) * TSEQ * HD;
  const float* wr = W + (long)j * HD;
  const float* wz = W + (long)(HD + j) * HD;
  const float* wn = W + (long)(2 * HD + j) * HD;
  const float br = bh[j], bz_ = bh[HD + j], bn_ = bh[2 * HD + j];
  for (int t = 0; t < TSEQ; ++t) {
    const float* gt = g + (long)t * 3 * HD;
    float ar = br, az = bz_, an = bn_;
    for (int k = 0; k < HD; ++k) {
      float hk = h[k];
      ar = fmaf(wr[k], hk, ar);
      az = fmaf(wz[k], hk, az);
      an = fmaf(wn[k], hk, an);
    }
    float r  = 1.f / (1.f + __expf(-(gt[j] + ar)));
    float zg = 1.f / (1.f + __expf(-(gt[HD + j] + az)));
    float n  = tanhf(gt[2 * HD + j] + r * an);
    float hn = (1.f - zg) * n + zg * h[j];
    __syncthreads();
    h[j] = hn;
    __syncthreads();
    o[(long)t * HD + j] = hn;
  }
}

// ---------------------------------------------------------------------------
// Masked attention over valid prefix + concat of modality summaries.
// One block per batch row.
// ---------------------------------------------------------------------------
__global__ void attn_kernel(const float* __restrict__ hs,
                            const float* __restrict__ att_w,
                            const int* __restrict__ seq_lens,
                            float* __restrict__ hcat)
{
  const int b = blockIdx.x, j = threadIdx.x;
  __shared__ float logit[3][TSEQ];
  __shared__ float alpha[3][TSEQ];
  const int L = seq_lens[b];
  for (int p = j; p < 3 * TSEQ; p += 128) {
    int m = p / TSEQ, t = p % TSEQ;
    const float* hrow = hs + (((long)m * NB + b) * TSEQ + t) * HD;
    const float* aw = att_w + m * HD;
    float s = 0.f;
    for (int k = 0; k < HD; ++k) s = fmaf(hrow[k], aw[k], s);
    logit[m][t] = s;
  }
  __syncthreads();
  if (j < 3) {
    int m = j;
    float mx = -__builtin_inff();
    for (int t = 0; t < L; ++t) mx = fmaxf(mx, logit[m][t]);
    float sum = 0.f;
    for (int t = 0; t < TSEQ; ++t) {
      float a = (t < L) ? __expf(logit[m][t] - mx) : 0.f;
      alpha[m][t] = a; sum += a;
    }
    float inv = 1.f / sum;
    for (int t = 0; t < TSEQ; ++t) alpha[m][t] *= inv;
  }
  __syncthreads();
  for (int m = 0; m < 3; ++m) {
    const float* base = hs + (((long)m * NB + b) * TSEQ) * HD + j;
    float s = 0.f;
    for (int t = 0; t < TSEQ; ++t) s = fmaf(alpha[m][t], base[(long)t * HD], s);
    hcat[(long)b * 3 * HD + m * HD + j] = s;
  }
}

// ---------------------------------------------------------------------------
// LayerNorm over HD. One block per batch row.
// ---------------------------------------------------------------------------
__global__ void ln_kernel(const float* __restrict__ x,
                          const float* __restrict__ g,
                          const float* __restrict__ be,
                          float* __restrict__ out)
{
  const int b = blockIdx.x, j = threadIdx.x;
  __shared__ float red[HD];
  float v = x[(long)b * HD + j];
  red[j] = v; __syncthreads();
  for (int s = 64; s > 0; s >>= 1) { if (j < s) red[j] += red[j + s]; __syncthreads(); }
  float mean = red[0] / HD;
  __syncthreads();
  float d = v - mean;
  red[j] = d * d; __syncthreads();
  for (int s = 64; s > 0; s >>= 1) { if (j < s) red[j] += red[j + s]; __syncthreads(); }
  float var = red[0] / HD;
  out[(long)b * HD + j] = g[j] * d * rsqrtf(var + 1e-5f) + be[j];
}

// ---------------------------------------------------------------------------
// Reparameterization: z = mu + exp(0.5*lv)*eps ; also pred_z[t=0].
// ---------------------------------------------------------------------------
__global__ void vae_kernel(const float* __restrict__ mu,
                           const float* __restrict__ lv,
                           const float* __restrict__ eps,
                           float* __restrict__ z,
                           float* __restrict__ predz)
{
  const int b = blockIdx.x, j = threadIdx.x;
  float m = mu[(long)b * HD + j], l = lv[(long)b * HD + j];
  float zz = m + __expf(0.5f * l) * eps[(long)b * HD + j];
  z[(long)b * HD + j] = zz;
  predz[(long)b * HD + j] = zz;   // slot (tout=0, b)
}

// Deterministic single-block ELBO reduction.
__global__ void elbo_kernel(const float* __restrict__ mu,
                            const float* __restrict__ lv,
                            float* __restrict__ out)
{
  __shared__ float red[256];
  const int tid = threadIdx.x;
  float acc = 0.f;
  for (int i = tid; i < NB * HD; i += 256) {
    float m = mu[i], l = lv[i];
    acc += 1.f + l - m * m - __expf(l);
  }
  red[tid] = acc; __syncthreads();
  for (int s = 128; s > 0; s >>= 1) { if (tid < s) red[tid] += red[tid + s]; __syncthreads(); }
  if (tid == 0) out[0] = -0.5f * red[0] / NB;
}

// ---------------------------------------------------------------------------
// GRU-ODE RK4 integrator, persistent per batch row (128 threads = HD lanes).
// ---------------------------------------------------------------------------
__device__ __forceinline__ float odef_eval(const float* __restrict__ wz,
                                           const float* __restrict__ wn,
                                           float* ys, float* zy, int j)
{
  __syncthreads();              // ys freshly written by all lanes
  float d1 = 0.f;
  for (int k = 0; k < HD; ++k) d1 = fmaf(wz[k], ys[k], d1);
  float zg = 1.f / (1.f + __expf(-d1));
  zy[j] = zg * ys[j];
  __syncthreads();
  float d2 = 0.f;
  for (int k = 0; k < HD; ++k) d2 = fmaf(wn[k], zy[k], d2);
  float n = tanhf(d2);
  float f = (1.f - zg) * (n - ys[j]);
  __syncthreads();              // safe to overwrite ys after this
  return f;
}

__global__ void ode_kernel(const float* __restrict__ z,
                           const float* __restrict__ hzW,
                           const float* __restrict__ hnW,
                           float* __restrict__ predz)
{
  const int b = blockIdx.x, j = threadIdx.x;
  __shared__ float ys[HD], zy[HD];
  float y = z[(long)b * HD + j];
  const float* wz = hzW + (long)j * HD;
  const float* wn = hnW + (long)j * HD;
  for (int it = 1; it < TOUT; ++it) {
    for (int s = 0; s < NSUB; ++s) {
      ys[j] = y;                         float f1 = odef_eval(wz, wn, ys, zy, j);
      ys[j] = y + 0.5f * DTs * f1;       float f2 = odef_eval(wz, wn, ys, zy, j);
      ys[j] = y + 0.5f * DTs * f2;       float f3 = odef_eval(wz, wn, ys, zy, j);
      ys[j] = y + DTs * f3;              float f4 = odef_eval(wz, wn, ys, zy, j);
      y += (DTs / 6.0f) * (f1 + 2.f * f2 + 2.f * f3 + f4);
    }
    predz[((long)it * NB + b) * HD + j] = y;
  }
}

// Row softmax over NCLS; rows are (tout*NB + b); output [B, TOUT, NCLS].
__global__ void softmax_kernel(const float* __restrict__ logits,
                               float* __restrict__ out)
{
  const int row = blockIdx.x, j = threadIdx.x;
  const int tout = row / NB, b = row % NB;
  __shared__ float red[NCLS];
  float v = logits[(long)row * NCLS + j];
  red[j] = v; __syncthreads();
  for (int s = 64; s > 0; s >>= 1) { if (j < s) red[j] = fmaxf(red[j], red[j + s]); __syncthreads(); }
  float mx = red[0]; __syncthreads();
  float e = __expf(v - mx);
  red[j] = e; __syncthreads();
  for (int s = 64; s > 0; s >>= 1) { if (j < s) red[j] += red[j + s]; __syncthreads(); }
  out[((long)b * TOUT + tout) * NCLS + j] = e / red[0];
}

// ---------------------------------------------------------------------------
// Host-side orchestration
// ---------------------------------------------------------------------------
extern "C" void kernel_launch(void* const* d_in, const int* in_sizes, int n_in,
                              void* d_out, int out_size, void* d_ws, size_t ws_size,
                              hipStream_t stream)
{
  const float* Hs_diag = (const float*)d_in[0];
  const float* Hs_drug = (const float*)d_in[1];
  const float* Hs_proc = (const float*)d_in[2];
  // d_in[3] = timestamps (unused: uniform unit spacing baked into NSUB/DT)
  const int*   seq_lens = (const int*)d_in[4];
  const float* X_d = (const float*)d_in[5];
  const float* X_m = (const float*)d_in[6];
  const float* X_p = (const float*)d_in[7];
  const float* W_ih = (const float*)d_in[8];
  const float* W_hh = (const float*)d_in[9];
  const float* b_ih = (const float*)d_in[10];
  const float* b_hh = (const float*)d_in[11];
  const float* att_w = (const float*)d_in[12];
  const float* fus_W1 = (const float*)d_in[13];
  const float* fus_b1 = (const float*)d_in[14];
  const float* fus_W2 = (const float*)d_in[15];
  const float* fus_b2 = (const float*)d_in[16];
  const float* ln_g = (const float*)d_in[17];
  const float* ln_b = (const float*)d_in[18];
  const float* mu_W = (const float*)d_in[19];
  const float* mu_b = (const float*)d_in[20];
  const float* var_W = (const float*)d_in[21];
  const float* var_b = (const float*)d_in[22];
  const float* hz_W = (const float*)d_in[23];
  const float* hn_W = (const float*)d_in[24];
  const float* fin_W = (const float*)d_in[25];
  const float* fin_b = (const float*)d_in[26];
  const float* eps = (const float*)d_in[27];

  float* out = (float*)d_out;
  float* out_pred = out;                          // [B, TOUT, NCLS]
  float* out_mu   = out + (long)NB * TOUT * NCLS; // [B, HD]
  float* out_lv   = out_mu + (long)NB * HD;       // [B, HD]
  float* out_elbo = out_lv + (long)NB * HD;       // scalar

  // workspace carve-out (floats)
  float* ws      = (float*)d_ws;
  float* emb_ws  = ws;                                   // 3*B*T*EMB
  float* gi_ws   = emb_ws + (long)3 * NB * TSEQ * EMBD;  // 3*B*T*384
  float* hs_ws   = gi_ws + (long)3 * NB * TSEQ * 3 * HD; // 3*B*T*HD
  float* hcat_ws = hs_ws + (long)3 * NB * TSEQ * HD;     // B*384
  float* h1_ws   = hcat_ws + (long)NB * 3 * HD;          // B*256
  float* h2_ws   = h1_ws + (long)NB * 2 * HD;            // B*128
  float* hln_ws  = h2_ws + (long)NB * HD;                // B*128
  float* z_ws    = hln_ws + (long)NB * HD;               // B*128
  float* predz   = z_ws + (long)NB * HD;                 // TOUT*B*HD
  float* logits  = predz + (long)TOUT * NB * HD;         // TOUT*B*NCLS

  // 1) Visit embeddings: emb[m,b,t,e] = sum_v Hs[b,v,t] * X[v,e]   (WMMA+TDM)
  //    A[t,v] strides: row(t)=1, col(v)=TSEQ ; batch = V*TSEQ per b.
  {
    const float* Hm[3] = {Hs_diag, Hs_drug, Hs_proc};
    const float* Xm[3] = {X_d, X_m, X_p};
    const int Vm[3] = {VD, VM, VP};
    for (int m = 0; m < 3; ++m) {
      gemm_wmma<<<dim3(1, 1, NB), 256, 0, stream>>>(
          Hm[m], (long)Vm[m] * TSEQ, 1, TSEQ,
          Xm[m], 0L, EMBD, 1,
          nullptr, 0,
          emb_ws + (long)m * NB * TSEQ * EMBD, (long)TSEQ * EMBD, EMBD,
          TSEQ, EMBD, Vm[m], 0);
    }
  }

  // 2) gi[m,b,t,g] = emb @ W_ih[m]^T + b_ih[m]   (M=B*T rows, N=384, K=128)
  gemm_wmma<<<dim3((NB * TSEQ) / MT, (3 * HD) / NTL, 3), 256, 0, stream>>>(
      emb_ws, (long)NB * TSEQ * EMBD, EMBD, 1,
      W_ih, (long)3 * HD * EMBD, 1, EMBD,
      b_ih, 3 * HD,
      gi_ws, (long)NB * TSEQ * 3 * HD, 3 * HD,
      NB * TSEQ, 3 * HD, EMBD, 0);

  // 3) GRU sequential scan over T
  gru_scan_kernel<<<dim3(NB, 3), 128, 0, stream>>>(gi_ws, W_hh, b_hh, hs_ws);

  // 4) Masked attention + concat
  attn_kernel<<<NB, 128, 0, stream>>>(hs_ws, att_w, seq_lens, hcat_ws);

  // 5) Fusion MLP (ReLU then linear), LayerNorm
  gemm_wmma<<<dim3(NB / MT, (2 * HD) / NTL, 1), 256, 0, stream>>>(
      hcat_ws, 0L, 3 * HD, 1, fus_W1, 0L, 1, 3 * HD, fus_b1, 0,
      h1_ws, 0L, 2 * HD, NB, 2 * HD, 3 * HD, 1);
  gemm_wmma<<<dim3(NB / MT, 1, 1), 256, 0, stream>>>(
      h1_ws, 0L, 2 * HD, 1, fus_W2, 0L, 1, 2 * HD, fus_b2, 0,
      h2_ws, 0L, HD, NB, HD, 2 * HD, 0);
  ln_kernel<<<NB, HD, 0, stream>>>(h2_ws, ln_g, ln_b, hln_ws);

  // 6) VAE heads straight into d_out
  gemm_wmma<<<dim3(NB / MT, 1, 1), 256, 0, stream>>>(
      hln_ws, 0L, HD, 1, mu_W, 0L, 1, HD, mu_b, 0,
      out_mu, 0L, HD, NB, HD, HD, 0);
  gemm_wmma<<<dim3(NB / MT, 1, 1), 256, 0, stream>>>(
      hln_ws, 0L, HD, 1, var_W, 0L, 1, HD, var_b, 0,
      out_lv, 0L, HD, NB, HD, HD, 0);

  // 7) Reparameterize + pred_z[0], ELBO
  vae_kernel<<<NB, HD, 0, stream>>>(out_mu, out_lv, eps, z_ws, predz);
  elbo_kernel<<<1, 256, 0, stream>>>(out_mu, out_lv, out_elbo);

  // 8) GRU-ODE RK4 trajectory (t = 1..7)
  ode_kernel<<<NB, HD, 0, stream>>>(z_ws, hz_W, hn_W, predz);

  // 9) Classifier logits + softmax into d_out
  gemm_wmma<<<dim3((TOUT * NB) / MT, 1, 1), 256, 0, stream>>>(
      predz, 0L, HD, 1, fin_W, 0L, 1, HD, fin_b, 0,
      logits, 0L, NCLS, TOUT * NB, NCLS, HD, 0);
  softmax_kernel<<<TOUT * NB, NCLS, 0, stream>>>(logits, out_pred);

  (void)in_sizes; (void)n_in; (void)out_size; (void)ws_size;
}
